// LiLoRALinearLayer_79465484910613
// MI455X (gfx1250) — matrix operand order repaired and checked by reading, not compiled, not run on hardware
//
#include <hip/hip_runtime.h>
#include <stdint.h>

// ---------------------------------------------------------------------------
// LiLoRA linear: out = X @ (W + up_aux@lora_up@lora_down@down_aux)^T + b
//   X: [65536, 1024] f32, W_eff: [1024, 1024], out: [65536, 1024] f32
// Strategy: fold weights once (tiny), then one bf16-WMMA GEMM with f32 acc.
// ---------------------------------------------------------------------------

typedef __bf16 bf16_t;
typedef __bf16 v4bf  __attribute__((ext_vector_type(4)));
typedef __bf16 v8bf  __attribute__((ext_vector_type(8)));
typedef __bf16 v16bf __attribute__((ext_vector_type(16)));
typedef float  v8f   __attribute__((ext_vector_type(8)));
typedef int    v4i_g __attribute__((vector_size(4 * sizeof(int))));  // matches builtin

#define M_TOTAL 65536   // B*S = 8*8192
#define N_TOTAL 1024    // OUT
#define K_TOTAL 1024    // IN
#define RANK    4
#define DDIM    100
#define UDIM    50
#define ALPHA_F 1.0f

#define BM 64           // block rows; block covers full N -> X read exactly once
#define BK 32           // K-step == WMMA K
#define LDA 40          // padded LDS row stride in bf16 (32 + 8), keeps 16B align
#define GEMM_THREADS 512  // 16 wave32 waves, each owns a 64x64 output sub-tile

// ---- optional gfx1250 async global->LDS path (probe via __has_builtin) ----
#if defined(__has_builtin)
# if __has_builtin(__builtin_amdgcn_global_load_async_to_lds_b128) && \
     __has_builtin(__builtin_amdgcn_s_wait_asynccnt)
#  define HAVE_ASYNC 1
# endif
#endif
#ifndef HAVE_ASYNC
# define HAVE_ASYNC 0
#endif

#if HAVE_ASYNC
__device__ __forceinline__ void async_cp16(bf16_t* dst_lds, const bf16_t* src_glb) {
  // signature (probed from diagnostics): 4 args, param0 = generic v4i* (src)
  __builtin_amdgcn_global_load_async_to_lds_b128(
      (v4i_g*)(void*)(uintptr_t)src_glb,
      (v4i_g*)(void*)(uintptr_t)dst_lds,
      0, 0);
}
#endif

// ---------------------------------------------------------------------------
// Prep 1: T2 = lora_up @ (lora_down @ down_aux)   -> [50][1024] f32
// ---------------------------------------------------------------------------
__global__ void lilora_chain_kernel(const float* __restrict__ lora_down, // [4][100]
                                    const float* __restrict__ lora_up,   // [50][4]
                                    const float* __restrict__ down_aux,  // [100][1024]
                                    float* __restrict__ T2)              // [50][1024]
{
  int tid = blockIdx.x * blockDim.x + threadIdx.x;
  if (tid >= UDIM * K_TOTAL) return;
  int u = tid >> 10;
  int k = tid & (K_TOTAL - 1);
  float acc = 0.f;
#pragma unroll
  for (int r = 0; r < RANK; ++r) {
    float t1 = 0.f;
    for (int d = 0; d < DDIM; ++d)
      t1 = fmaf(lora_down[r * DDIM + d], down_aux[(size_t)d * K_TOTAL + k], t1);
    acc = fmaf(lora_up[u * RANK + r], t1, acc);
  }
  T2[tid] = acc;
}

// ---------------------------------------------------------------------------
// Prep 2: W_bf16[n][k] = bf16( W[n][k] + ALPHA * sum_u up_aux[n][u]*T2[u][k] )
// ---------------------------------------------------------------------------
__global__ void fold_weight_kernel(const float* __restrict__ W,       // [1024][1024]
                                   const float* __restrict__ up_aux,  // [1024][50]
                                   const float* __restrict__ T2,      // [50][1024]
                                   bf16_t* __restrict__ Wb)           // [1024][1024]
{
  int tid = blockIdx.x * blockDim.x + threadIdx.x;
  int n = tid >> 10;
  int k = tid & (K_TOTAL - 1);
  float acc = 0.f;
  for (int u = 0; u < UDIM; ++u)
    acc = fmaf(up_aux[n * UDIM + u], T2[(size_t)u * K_TOTAL + k], acc);
  Wb[tid] = (bf16_t)(W[tid] + ALPHA_F * acc);
}

// ---------------------------------------------------------------------------
// Main GEMM: Y[m,n] = sum_k X[m,k]*Weff[n,k] + b[n]
// 512 threads = 16 waves. Block tile 64 x 1024; wave tile 64 x 64 (4x4 WMMA).
// LDS: As = 64x32 bf16 (converted from f32), Bs = 1024x32 bf16, both row-padded.
// ---------------------------------------------------------------------------
__launch_bounds__(GEMM_THREADS)
__global__ void lilora_gemm_kernel(const float*  __restrict__ X,
                                   const bf16_t* __restrict__ Wb,
                                   const float*  __restrict__ bias,
                                   float*        __restrict__ out)
{
  __shared__ bf16_t As[BM * LDA];        // 5  KB
  __shared__ bf16_t Bs[N_TOTAL * LDA];   // 80 KB

  const int tid  = threadIdx.x;
  const int wave = tid >> 5;
  const int lane = tid & 31;
  const int lm   = lane & 15;   // row/col within 16
  const int hi   = lane >> 4;   // half-wave select
  const int m0   = blockIdx.x * BM;
  const int nw   = wave * 64;   // this wave's N base

  const v8f vzero = {0.f, 0.f, 0.f, 0.f, 0.f, 0.f, 0.f, 0.f};
  v8f acc[4][4];
#pragma unroll
  for (int tm = 0; tm < 4; ++tm)
#pragma unroll
    for (int tn = 0; tn < 4; ++tn)
      acc[tm][tn] = vzero;

  for (int k0 = 0; k0 < K_TOTAL; k0 += BK) {
    // ---- stage A tile: 64 rows x 32 f32 -> bf16 in LDS (one float4 / thread)
    {
      const int row = tid >> 3;
      const int c4  = (tid & 7) << 2;
      const float4 f =
          *(const float4*)(X + (size_t)(m0 + row) * K_TOTAL + k0 + c4);
      v4bf p;
      p[0] = (bf16_t)f.x; p[1] = (bf16_t)f.y;
      p[2] = (bf16_t)f.z; p[3] = (bf16_t)f.w;
      *(v4bf*)(&As[row * LDA + c4]) = p;
    }
    // ---- stage B panel: 1024 rows x 32 bf16 (8 x 16B chunks / thread)
#pragma unroll
    for (int i = 0; i < 8; ++i) {
      const int c   = tid + i * GEMM_THREADS;  // 0..4095 chunk id
      const int row = c >> 2;
      const int cc  = (c & 3) << 3;            // bf16 offset within row
      const bf16_t* src = Wb + (size_t)row * K_TOTAL + k0 + cc;
      bf16_t*       dst = &Bs[row * LDA + cc];
#if HAVE_ASYNC
      async_cp16(dst, src);                    // global_load_async_to_lds_b128
#else
      *(v8bf*)dst = *(const v8bf*)src;
#endif
    }
#if HAVE_ASYNC
    __builtin_amdgcn_s_wait_asynccnt(0);
#endif
    __syncthreads();

    // ---- load fragments per documented 16-bit WMMA VGPR layouts
    v16bf a[4], b[4];
#pragma unroll
    for (int tm = 0; tm < 4; ++tm) {
      // A: lane holds row m = lm, K = hi*8 + {0..7} then 16 + hi*8 + {0..7}
      const bf16_t* p = &As[(tm * 16 + lm) * LDA + hi * 8];
      v8bf lo = *(const v8bf*)p;
      v8bf hv = *(const v8bf*)(p + 16);
      a[tm] = __builtin_shufflevector(lo, hv, 0, 1, 2, 3, 4, 5, 6, 7,
                                      8, 9, 10, 11, 12, 13, 14, 15);
    }
#pragma unroll
    for (int tn = 0; tn < 4; ++tn) {
      // B: lane holds column n = lm, K = hi*8 + {0..7} then 16 + hi*8 + {0..7}
      const bf16_t* p = &Bs[(nw + tn * 16 + lm) * LDA + hi * 8];
      v8bf lo = *(const v8bf*)p;
      v8bf hv = *(const v8bf*)(p + 16);
      b[tn] = __builtin_shufflevector(lo, hv, 0, 1, 2, 3, 4, 5, 6, 7,
                                      8, 9, 10, 11, 12, 13, 14, 15);
    }
    // ---- 16 WMMAs per wave per K-step
#pragma unroll
    for (int tm = 0; tm < 4; ++tm)
#pragma unroll
      for (int tn = 0; tn < 4; ++tn)
        acc[tm][tn] = __builtin_amdgcn_wmma_f32_16x16x32_bf16(
            false, a[tm], false, b[tn], (short)0, acc[tm][tn], false, false);

    __syncthreads();
  }

  // ---- epilogue: C/D layout -> VGPR v holds row (v + 8*hi), col lm
#pragma unroll
  for (int tn = 0; tn < 4; ++tn) {
    const int n  = nw + tn * 16 + lm;
    const float bn = bias[n];
#pragma unroll
    for (int tm = 0; tm < 4; ++tm) {
#pragma unroll
      for (int v = 0; v < 8; ++v) {
        const int m = m0 + tm * 16 + hi * 8 + v;
        out[(size_t)m * N_TOTAL + n] = acc[tm][tn][v] + bn;
      }
    }
  }
}

// ---------------------------------------------------------------------------
extern "C" void kernel_launch(void* const* d_in, const int* in_sizes, int n_in,
                              void* d_out, int out_size, void* d_ws, size_t ws_size,
                              hipStream_t stream) {
  (void)in_sizes; (void)n_in; (void)out_size; (void)ws_size;

  const float* X         = (const float*)d_in[0];  // [8,8192,1024]
  const float* W         = (const float*)d_in[1];  // [1024,1024]
  const float* b         = (const float*)d_in[2];  // [1024]
  const float* lora_down = (const float*)d_in[3];  // [4,100]
  const float* lora_up   = (const float*)d_in[4];  // [50,4]
  const float* down_aux  = (const float*)d_in[5];  // [100,1024]
  const float* up_aux    = (const float*)d_in[6];  // [1024,50]
  float* out = (float*)d_out;

  // workspace: [0, 2MB) W_eff in bf16; then T2 [50][1024] f32 (~200 KB)
  bf16_t* Wb = (bf16_t*)d_ws;
  float*  T2 = (float*)((char*)d_ws + (size_t)N_TOTAL * K_TOTAL * sizeof(bf16_t));

  lilora_chain_kernel<<<(UDIM * K_TOTAL + 255) / 256, 256, 0, stream>>>(
      lora_down, lora_up, down_aux, T2);
  fold_weight_kernel<<<(N_TOTAL * K_TOTAL) / 256, 256, 0, stream>>>(
      W, up_aux, T2, Wb);
  lilora_gemm_kernel<<<M_TOTAL / BM, GEMM_THREADS, 0, stream>>>(X, Wb, b, out);
}
